// Res_NL_18751827214433
// MI455X (gfx1250) — matrix operand order repaired and checked by reading, not compiled, not run on hardware
//
#include <hip/hip_runtime.h>
#include <hip/hip_bf16.h>
#include <math.h>

// ---------------------------------------------------------------------------
// Non-local attention block, MI455X (gfx1250), wave32 + WMMA bf16 + TDM.
//   b=8, c=256, h=w=64  ->  HW=4096, inter=128, N=2048 (buffer-reinterpret)
// Pipeline:
//   K1  phi/theta/g projections (fp32 -> bf16). phiT/thetaT get the
//       (128,4096)->(2048,256) view-transpose fused into the store (so the
//       big GEMMs see K-contiguous operands); g stays (256,2048) c-major.
//   K2  attn = thetaT x phiT^T  via LDS kernel: TDM tensor_load_to_lds
//       double-buffered tiles + ds_load_b128 fragments + WMMA.
//   K3  column softmax stats -> s[j] = exp(-m[j]) / sum_i exp(a[i,j]-m[j])
//   K4  fold s[j] into g (GEMM-5's A side then only needs a uniform exp()).
//   K5  y = exp(attn) x g'   (LDS/TDM kernel), transposed store.
//   K6  mask = w_mask x yView + fused tanh(alpha) residual (fp32 out).
// Workspace: ~100.8 MB carved from d_ws.
// ---------------------------------------------------------------------------

typedef __attribute__((ext_vector_type(16))) __bf16 v16bf;
typedef __attribute__((ext_vector_type(8)))  __bf16 v8bf;
typedef __attribute__((ext_vector_type(8)))  float  v8f;
typedef __attribute__((ext_vector_type(4)))  unsigned int u32x4;
typedef __attribute__((ext_vector_type(8)))  int i32x8;
typedef __attribute__((ext_vector_type(4)))  int i32x4;

#define STORE_PLAIN 0
#define STORE_TPROJ 1   // dst = (q % 2048)*256 + q/2048, q = row*4096+col
#define STORE_TSWAP 2   // dst = col*2048 + row
#define STORE_EPI   3   // out = tanh(alpha)*(acc + x) + x  (fp32)

// ---------------------------------------------------------------------------
// TDM: issue a 2-D tile load global->LDS. D# per CDNA5 ISA ch.8 (08_async_
// tensor.md): group0 = {count/type/lds_addr/global_addr}, group1 = dims.
// bf16 elements (data_size code 1 = 2 bytes). Rows of the tile are
// K-contiguous in memory with stride rowStride (elements).
// ---------------------------------------------------------------------------
__device__ __forceinline__ void tdm_load_2d(unsigned ldsAddr, const void* gptr,
                                            unsigned tileK, unsigned tileRows,
                                            unsigned long long rowStride,
                                            unsigned tensorRows)
{
    unsigned long long ga = (unsigned long long)(size_t)gptr;
    u32x4 g0;
    g0[0] = 1u;                                   // count=1, user descriptor
    g0[1] = ldsAddr;                              // lds_addr (bytes)
    g0[2] = (unsigned)ga;                         // global_addr[31:0]
    g0[3] = (unsigned)(ga >> 32) | 0x80000000u;   // addr[56:32] | type=2

    unsigned long long strideBits = rowStride;    // tensor_dim0_stride (elems)
    unsigned tdim0 = (unsigned)rowStride;         // tensor_dim0 >= tile span
    i32x8 g1;
    g1[0] = (int)(1u << 16);                              // data_size=2B
    g1[1] = (int)((tdim0 & 0xFFFFu) << 16);               // tensor_dim0 lo
    g1[2] = (int)(((tdim0 >> 16) & 0xFFFFu) |
                  ((tensorRows & 0xFFFFu) << 16));        // td0 hi | td1 lo
    g1[3] = (int)(((tensorRows >> 16) & 0xFFFFu) |
                  ((tileK & 0xFFFFu) << 16));             // td1 hi | tile_dim0
    g1[4] = (int)(tileRows & 0xFFFFu);                    // tile_dim1 (dim2=0)
    g1[5] = (int)(unsigned)(strideBits & 0xFFFFFFFFull);  // stride0[31:0]
    g1[6] = (int)(unsigned)((strideBits >> 32) & 0xFFFFull); // stride0[47:32]
    g1[7] = 0;

    i32x4 z4 = {0, 0, 0, 0};
#if defined(__clang_major__) && (__clang_major__ >= 23)
    i32x8 z8 = {0, 0, 0, 0, 0, 0, 0, 0};
    __builtin_amdgcn_tensor_load_to_lds(g0, g1, z4, z4, z8, 0);
#else
    __builtin_amdgcn_tensor_load_to_lds(g0, g1, z4, z4, 0);
#endif
}

// Build an A fragment (16x32 bf16, ISA 7.12.2 layout) from an LDS tile stored
// row-major M x 32 (K contiguous): per lane two contiguous 8-element runs.
template<bool EXPA>
__device__ __forceinline__ v16bf lds_a_frag(const __bf16* base, int mLocal, int half)
{
    const v8bf* p = (const v8bf*)(base + mLocal * 32 + half * 8);
    v8bf lo = p[0];            // K = half*8 .. +7
    v8bf hi = p[2];            // K = 16 + half*8 .. +7
    v16bf a;
    #pragma unroll
    for (int i = 0; i < 8; ++i) {
        if constexpr (EXPA) {
            a[i]     = (__bf16)__expf((float)lo[i]);
            a[8 + i] = (__bf16)__expf((float)hi[i]);
        } else {
            a[i] = lo[i]; a[8 + i] = hi[i];
        }
    }
    return a;
}

// ---------------------------------------------------------------------------
// LDS/TDM GEMM:  D(M,N) = A(M,K) x BT(N,K)^T, all bf16, f32 accumulate.
// A row-major (lda, K contiguous); BT row-major (ldbt, K contiguous).
// Block tile 64x256 (8 waves, wave tile 32x64), K-step 32, double-buffered
// TDM tiles: A 64x32 (4KB), B 256x32 (16KB).
// ---------------------------------------------------------------------------
template<bool EXPA, int SMODE>
__global__ __launch_bounds__(256) void gemm_wmma_lds(
    const __bf16* __restrict__ A,  long long aBatch, int lda,
    const __bf16* __restrict__ BT, long long bBatch, int ldbt,
    __bf16*       __restrict__ D,  long long dBatch, int ldd,
    int M, int N, int K)
{
    __shared__ __align__(32) __bf16 Alds[2][64 * 32];
    __shared__ __align__(32) __bf16 Blds[2][256 * 32];

    const int tid  = threadIdx.x;
    const int wave = tid >> 5;
    const int lane = tid & 31;
    const int half = lane >> 4;
    const int l16  = lane & 15;
    const int z    = blockIdx.z;

    const int mBlk = blockIdx.y * 64;
    const int nBlk = blockIdx.x * 256;
    const int mW   = (wave >> 2) * 32;   // wave row offset in block tile
    const int nW   = (wave & 3) * 64;    // wave col offset in block tile

    const __bf16* Ab = A  + (long long)z * aBatch;
    const __bf16* Bb = BT + (long long)z * bBatch;
    __bf16*       Dp = D  + (long long)z * dBatch;

    v8f acc[2][4] = {};
    const int numK = K >> 5;

    if (wave == 0) {   // prologue: TDM loads for buffer 0 (EXEC ignored by TDM)
        tdm_load_2d((unsigned)(size_t)&Alds[0][0],
                    Ab + (long long)mBlk * lda, 32, 64,
                    (unsigned long long)lda, (unsigned)M);
        tdm_load_2d((unsigned)(size_t)&Blds[0][0],
                    Bb + (long long)nBlk * ldbt, 32, 256,
                    (unsigned long long)ldbt, (unsigned)N);
    }

    for (int it = 0; it < numK; ++it) {
        const int cur = it & 1;
        if (wave == 0) {
            if (it + 1 < numK) {
                const int nxt = cur ^ 1, k0n = (it + 1) << 5;
                tdm_load_2d((unsigned)(size_t)&Alds[nxt][0],
                            Ab + (long long)mBlk * lda + k0n, 32, 64,
                            (unsigned long long)lda, (unsigned)M);
                tdm_load_2d((unsigned)(size_t)&Blds[nxt][0],
                            Bb + (long long)nBlk * ldbt + k0n, 32, 256,
                            (unsigned long long)ldbt, (unsigned)N);
                __builtin_amdgcn_s_wait_tensorcnt(2);  // cur pair complete
            } else {
                __builtin_amdgcn_s_wait_tensorcnt(0);
            }
        }
        __syncthreads();   // cur tiles visible to all waves

        v16bf afr[2];
        #pragma unroll
        for (int fm = 0; fm < 2; ++fm)
            afr[fm] = lds_a_frag<EXPA>(&Alds[cur][0], mW + fm * 16 + l16, half);

        v16bf bfr[4];
        #pragma unroll
        for (int fn = 0; fn < 4; ++fn)
        {
            const v8bf* pb = (const v8bf*)(&Blds[cur][0] +
                                           (nW + fn * 16 + l16) * 32 + half * 8);
            v8bf lo = pb[0], hi = pb[2];
            #pragma unroll
            for (int i = 0; i < 8; ++i) { bfr[fn][i] = lo[i]; bfr[fn][8 + i] = hi[i]; }
        }

        #pragma unroll
        for (int fm = 0; fm < 2; ++fm)
            #pragma unroll
            for (int fn = 0; fn < 4; ++fn)
                acc[fm][fn] = __builtin_amdgcn_wmma_f32_16x16x32_bf16(
                    false, afr[fm], false, bfr[fn],
                    (short)0, acc[fm][fn], false, false);

        __syncthreads();   // all waves done with cur before TDM overwrites it
    }

    // f32 C/D layout: VGPR r -> M = r + 8*half, N = l16
    #pragma unroll
    for (int fm = 0; fm < 2; ++fm) {
        #pragma unroll
        for (int fn = 0; fn < 4; ++fn) {
            const int col = nBlk + nW + fn * 16 + l16;
            #pragma unroll
            for (int r = 0; r < 8; ++r) {
                const int row = mBlk + mW + fm * 16 + r + half * 8;
                const float v = acc[fm][fn][r];
                if constexpr (SMODE == STORE_PLAIN)
                    Dp[(long long)row * ldd + col] = (__bf16)v;
                else // STORE_TSWAP
                    Dp[(long long)col * 2048 + row] = (__bf16)v;
            }
        }
    }
}

// ---------------------------------------------------------------------------
// Direct-load GEMM for the small fp32-A stages (projections + mask epilogue).
// ---------------------------------------------------------------------------
template<bool B_F32, int SMODE>
__global__ __launch_bounds__(256) void gemm_wmma_direct(
    const float* __restrict__ Aw, int lda,
    const void*  __restrict__ Bv, long long bBatch, int ldb,
    __bf16*      __restrict__ D,  long long dBatch, int ldd,
    int M, int N, int K,
    const float* __restrict__ xres, const float* __restrict__ alpha,
    float* __restrict__ outp, long long xBatch)
{
    const int tid  = threadIdx.x;
    const int wave = tid >> 5;
    const int lane = tid & 31;
    const int half = lane >> 4;
    const int l16  = lane & 15;
    const int z    = blockIdx.z;

    const int m0 = blockIdx.y * 64  + (wave >> 2) * 32;
    const int n0 = blockIdx.x * 256 + (wave &  3) * 64;
    if (m0 >= M || n0 >= N) return;

    const float*  Bf32 = (const float*)Bv  + (long long)z * bBatch;
    const __bf16* Bb   = (const __bf16*)Bv + (long long)z * bBatch;
    __bf16*       Dp   = D    + (long long)z * dBatch;
    const float*  Xp   = xres + (long long)z * xBatch;
    float*        Op   = outp + (long long)z * xBatch;

    float t = 0.0f;
    if constexpr (SMODE == STORE_EPI) t = tanhf(alpha[0]);

    v8f acc[2][4] = {};

    for (int k0 = 0; k0 < K; k0 += 32) {
        v16bf afr[2];
        #pragma unroll
        for (int fm = 0; fm < 2; ++fm) {
            const int row = m0 + fm * 16 + l16;
            #pragma unroll
            for (int p = 0; p < 8; ++p) {
                const int kk = ((p & 3) * 2) + ((p >> 2) * 16) + half * 8;
                const long long a0 = (long long)row * lda + (k0 + kk);
                afr[fm][2 * p]     = (__bf16)Aw[a0];
                afr[fm][2 * p + 1] = (__bf16)Aw[a0 + 1];
            }
        }
        v16bf bfr[4];
        #pragma unroll
        for (int fn = 0; fn < 4; ++fn) {
            const int col = n0 + fn * 16 + l16;
            #pragma unroll
            for (int p = 0; p < 8; ++p) {
                const int kk = ((p & 3) * 2) + ((p >> 2) * 16) + half * 8;
                const long long b0 = (long long)(k0 + kk) * ldb + col;
                if constexpr (B_F32) {
                    bfr[fn][2 * p]     = (__bf16)Bf32[b0];
                    bfr[fn][2 * p + 1] = (__bf16)Bf32[b0 + ldb];
                } else {
                    bfr[fn][2 * p]     = Bb[b0];
                    bfr[fn][2 * p + 1] = Bb[b0 + ldb];
                }
            }
        }
        if (k0 + 32 < K) {  // global_prefetch_b8
            if constexpr (B_F32)
                __builtin_prefetch(&Bf32[(long long)(k0 + 32) * ldb + n0 + l16], 0, 1);
            else
                __builtin_prefetch(&Bb[(long long)(k0 + 32) * ldb + n0 + l16], 0, 1);
        }
        #pragma unroll
        for (int fm = 0; fm < 2; ++fm)
            #pragma unroll
            for (int fn = 0; fn < 4; ++fn)
                acc[fm][fn] = __builtin_amdgcn_wmma_f32_16x16x32_bf16(
                    false, afr[fm], false, bfr[fn],
                    (short)0, acc[fm][fn], false, false);
    }

    #pragma unroll
    for (int fm = 0; fm < 2; ++fm) {
        #pragma unroll
        for (int fn = 0; fn < 4; ++fn) {
            const int col = n0 + fn * 16 + l16;
            #pragma unroll
            for (int r = 0; r < 8; ++r) {
                const int row = m0 + fm * 16 + r + half * 8;
                const float v = acc[fm][fn][r];
                if constexpr (SMODE == STORE_PLAIN) {
                    Dp[(long long)row * ldd + col] = (__bf16)v;
                } else if constexpr (SMODE == STORE_TPROJ) {
                    const long long q = (long long)row * ldd + col;
                    Dp[((q & 2047) << 8) + (q >> 11)] = (__bf16)v;
                } else { // STORE_EPI
                    const long long q = (long long)row * ldd + col;
                    const float xi = Xp[q];
                    Op[q] = t * (v + xi) + xi;
                }
            }
        }
    }
}

// Column softmax stats over attn (2048x2048/batch), softmax over rows i for
// each column j (torch Softmax(dim=1)).  s[j] = exp(-m[j]) / D[j].
__global__ __launch_bounds__(256) void softmax_colstats(
    const __bf16* __restrict__ attn, float* __restrict__ s)
{
    const int z = blockIdx.y;
    const int j = blockIdx.x * 256 + threadIdx.x;
    const __bf16* A = attn + (long long)z * 2048 * 2048;
    float m = -INFINITY;
    for (int i = 0; i < 2048; ++i)
        m = fmaxf(m, (float)A[(long long)i * 2048 + j]);
    float d = 0.0f;
    for (int i = 0; i < 2048; ++i)
        d += __expf((float)A[(long long)i * 2048 + j] - m);
    s[z * 2048 + j] = __expf(-m) / d;
}

// g'[c,j] = g[c,j] * s[j]   (g is (256,2048) c-major; rewritten every call)
__global__ __launch_bounds__(256) void scale_g_rows(
    __bf16* __restrict__ g, const float* __restrict__ s)
{
    const int z = blockIdx.y;
    const long long i = (long long)blockIdx.x * 256 + threadIdx.x; // < 524288
    const int j = (int)(i & 2047);
    __bf16* G = g + (long long)z * 524288;
    G[i] = (__bf16)((float)G[i] * s[z * 2048 + j]);
}

extern "C" void kernel_launch(void* const* d_in, const int* in_sizes, int n_in,
                              void* d_out, int out_size, void* d_ws, size_t ws_size,
                              hipStream_t stream) {
    const float* x       = (const float*)d_in[0]; // (8,256,64,64)
    const float* w_phi   = (const float*)d_in[1]; // (128,256)
    const float* w_theta = (const float*)d_in[2]; // (128,256)
    const float* w_g     = (const float*)d_in[3]; // (128,256)
    const float* w_mask  = (const float*)d_in[4]; // (256,128)
    const float* alpha   = (const float*)d_in[5]; // (1,)
    float* out = (float*)d_out;                   // (8,256,64,64)

    // --- workspace carve (assumes ws_size >= ~101 MB) ---
    char* w = (char*)d_ws;
    __bf16* phiT   = (__bf16*)(w);               //  8 x (2048,256) bf16
    __bf16* thetaT = (__bf16*)(w + 8388608);     //  8 x (2048,256)
    __bf16* gbuf   = (__bf16*)(w + 16777216);    //  8 x (256,2048) c-major
    __bf16* attn   = (__bf16*)(w + 25165824);    //  8 x (2048,2048) (67.1 MB)
    float*  s      = (float*) (w + 92274688);    //  8 x 2048 f32
    __bf16* ybuf   = (__bf16*)(w + 92340224);    //  8 x (256,2048) c-major

    const long long xB = 1048576;   // 256*4096 f32 per batch
    const long long pB = 524288;    // projection elems per batch
    const long long aB = 4194304;   // 2048*2048 per batch
    dim3 blk(256);

    // K1: projections  M=128, N=4096, K=256 (fp32 in, bf16 out)
    dim3 gProj(16, 2, 8);
    gemm_wmma_direct<true, STORE_TPROJ><<<gProj, blk, 0, stream>>>(
        w_phi, 256, x, xB, 4096, phiT, pB, 4096, 128, 4096, 256,
        nullptr, nullptr, nullptr, 0);
    gemm_wmma_direct<true, STORE_TPROJ><<<gProj, blk, 0, stream>>>(
        w_theta, 256, x, xB, 4096, thetaT, pB, 4096, 128, 4096, 256,
        nullptr, nullptr, nullptr, 0);
    gemm_wmma_direct<true, STORE_PLAIN><<<gProj, blk, 0, stream>>>(
        w_g, 256, x, xB, 4096, gbuf, pB, 4096, 128, 4096, 256,
        nullptr, nullptr, nullptr, 0);

    // K2: attn(2048,2048) = thetaT(2048,256) x phiT(2048,256)^T   [TDM+LDS]
    dim3 gAttn(8, 32, 8);
    gemm_wmma_lds<false, STORE_PLAIN><<<gAttn, blk, 0, stream>>>(
        thetaT, pB, 256, phiT, pB, 256, attn, aB, 2048, 2048, 2048, 256);

    // K3/K4: column softmax stats, fold into g
    softmax_colstats<<<dim3(8, 8), blk, 0, stream>>>(attn, s);
    scale_g_rows<<<dim3(2048, 8), blk, 0, stream>>>(gbuf, s);

    // K5: y = exp(attn)(2048,2048) x gbuf(256,2048)^T, transposed store [TDM+LDS]
    dim3 gY(1, 32, 8);
    gemm_wmma_lds<true, STORE_TSWAP><<<gY, blk, 0, stream>>>(
        attn, aB, 2048, gbuf, pB, 2048, ybuf, pB, 256, 2048, 256, 2048);

    // K6: mask GEMM + fused residual epilogue (M=256,N=4096,K=128, fp32 out)
    dim3 gMask(16, 4, 8);
    gemm_wmma_direct<false, STORE_EPI><<<gMask, blk, 0, stream>>>(
        w_mask, 128, ybuf, pB, 4096, nullptr, 0, 4096, 256, 4096, 128,
        x, alpha, out, xB);
}
